// DoubleControlCDE_50311246905433
// MI455X (gfx1250) — matrix operand order repaired and checked by reading, not compiled, not run on hardware
//
#include <hip/hip_runtime.h>
#include <hip/hip_bf16.h>
#include <math.h>

// ---------------- types ----------------
typedef __attribute__((ext_vector_type(16))) __bf16 v16bf;
typedef __attribute__((ext_vector_type(8)))  float  v8f;

// Problem constants (match reference)
#define BB   512
#define TT   512
#define LL   256
#define FF   256      // future steps = T - L
#define XC   16
#define HH   128
#define G3   384      // 3*H
#define OC   8
#define NZ   8
#define CF   9
#define WW   128

__device__ __forceinline__ float sigmoidf_(float x) { return 1.0f / (1.0f + __expf(-x)); }
__device__ __forceinline__ float softplusf_(float x) { return (x > 20.0f) ? x : log1pf(__expf(x)); }

__device__ __forceinline__ v8f wmma_bf16(v16bf a, v16bf b, v8f c) {
    return __builtin_amdgcn_wmma_f32_16x16x32_bf16(false, a, false, b, (short)0, c, false, false);
}

// A-fragment (16x32 bf16, M x K), contiguous K to match B:
// lanes 0-15: M=lane, K=0..15; lanes 16-31: M=lane-16, K=16..31
__device__ __forceinline__ v16bf load_a_frag(const __bf16* smem, int ld, int lane, int k0) {
    int m = lane & 15, half = lane >> 4;
    v16bf a;
#pragma unroll
    for (int e = 0; e < 16; ++e) {
        int k = (half << 4) + e;
        a[e] = smem[m * ld + k0 + k];
    }
    return a;
}

// B-fragment (32x16 bf16, K x N), K-major source with row stride ldn:
// lanes 0-15 hold K=0..15, lanes 16-31 hold K=16..31; n = lane&15
__device__ __forceinline__ v16bf load_b_frag(const __bf16* w, int ldn, int lane, int k0, int n0) {
    int n = lane & 15, half = lane >> 4;
    v16bf b;
#pragma unroll
    for (int e = 0; e < 16; ++e) {
        int k = (half << 4) + e;
        b[e] = w[(size_t)(k0 + k) * ldn + n0 + n];
    }
    return b;
}

// ---------------- kernel 0a: transpose fp32 weight (R rows, K cols) -> bf16 [Kpad][R] ----------------
__global__ void transpose_to_bf16(const float* __restrict__ src, __bf16* __restrict__ dst,
                                  int R, int K, int Kpad) {
    int i = blockIdx.x * blockDim.x + threadIdx.x;
    int total = Kpad * R;
    if (i >= total) return;
    int k = i / R, r = i % R;
    float v = (k < K) ? src[(size_t)r * K + k] : 0.0f;
    dst[i] = (__bf16)v;
}

// ---------------- kernel 0b: pack f_w3 (1152,128) into fragment-major bf16 ----------------
// dst index = (((tile*4 + ks)*32 + lane)*16 + e)
// fragment element (lane,e): k = ks*32 + (lane>>4)*16 + e ; n = tile*16 + (lane&15)
__global__ void pack_w3_frags(const float* __restrict__ src, __bf16* __restrict__ dst) {
    int i = blockIdx.x * blockDim.x + threadIdx.x;
    const int total = 72 * 4 * 32 * 16;
    if (i >= total) return;
    int e  = i & 15;
    int l  = (i >> 4) & 31;
    int ks = (i >> 9) & 3;
    int t  = i >> 11;
    int k = ks * 32 + ((l >> 4) << 4) + e;
    int n = t * 16 + (l & 15);
    dst[i] = (__bf16)src[(size_t)n * WW + k];
}

// ---------------- kernel 1: zx GRU over T steps + zx linear head ----------------
__global__ __launch_bounds__(256) void zx_gru_kernel(
    const float* __restrict__ coeffs_x, const float* __restrict__ tarr,
    const __bf16* __restrict__ wihT, const __bf16* __restrict__ whhT,
    const float* __restrict__ gb, const float* __restrict__ gbn,
    const float* __restrict__ lw, const float* __restrict__ lb,
    float* __restrict__ zx_out)
{
    __shared__ __bf16 xs[16][32];
    __shared__ __bf16 hbf[16][HH];
    __shared__ float  hf[16][HH];
    __shared__ float  srz[16][2 * HH];
    __shared__ float  sinn[16][HH];
    __shared__ float  sgnn[16][HH];

    const int b0 = blockIdx.x * 16;
    const int tid = threadIdx.x;
    const int wv = tid >> 5, lane = tid & 31;
    const int nn = lane & 15, half = lane >> 4;

    // Preload per-wave weight fragments: wave wv, gate j -> cols wv*16 + j*128
    v16bf bih[3], bhh[3][4];
    float bias[3];
#pragma unroll
    for (int j = 0; j < 3; ++j) {
        int n0 = wv * 16 + j * HH;
        bih[j] = load_b_frag(wihT, G3, lane, 0, n0);
#pragma unroll
        for (int ks = 0; ks < 4; ++ks) bhh[j][ks] = load_b_frag(whhT, G3, lane, ks * 32, n0);
        bias[j] = gb[n0 + nn];
    }

    for (int i = tid; i < 16 * HH; i += 256) {
        (&hf[0][0])[i] = 0.0f;
        (&hbf[0][0])[i] = (__bf16)0.0f;
    }
    float hreg[8];
#pragma unroll
    for (int e = 0; e < 8; ++e) hreg[e] = 0.0f;
    __syncthreads();

    for (int t = 0; t < TT; ++t) {
        // assemble x tile [16][32]: [coeffs_x(16), t, pad]
        for (int i = tid; i < 16 * 32; i += 256) {
            int r = i >> 5, c = i & 31;
            float v = 0.0f;
            if (c < XC) v = coeffs_x[((size_t)(b0 + r) * TT + t) * XC + c];
            else if (c == XC) v = tarr[t];
            (&xs[0][0])[i] = (__bf16)v;
        }
        __syncthreads();

        v16bf ax = load_a_frag(&xs[0][0], 32, lane, 0);
        v16bf ah[4];
#pragma unroll
        for (int ks = 0; ks < 4; ++ks) ah[ks] = load_a_frag(&hbf[0][0], HH, lane, ks * 32);

        v8f acc_i[3], acc_g[3];
#pragma unroll
        for (int j = 0; j < 3; ++j) {
            v8f zi = {};
            acc_i[j] = wmma_bf16(ax, bih[j], zi);
            v8f zg = {};
#pragma unroll
            for (int ks = 0; ks < 4; ++ks) zg = wmma_bf16(ah[ks], bhh[j][ks], zg);
            acc_g[j] = zg;
        }
        // phase 1: store gate pre-activations
#pragma unroll
        for (int v = 0; v < 8; ++v) {
            int row = v + (half << 3);
            int col = wv * 16 + nn;
            srz[row][col]      = acc_i[0][v] + acc_g[0][v] + bias[0];
            srz[row][HH + col] = acc_i[1][v] + acc_g[1][v] + bias[1];
            sinn[row][col]     = acc_i[2][v] + bias[2];
            sgnn[row][col]     = acc_g[2][v];
        }
        __syncthreads();
        // phase 2: gate math, update h
#pragma unroll
        for (int e = 0; e < 8; ++e) {
            int idx = tid + 256 * e;
            int row = idx >> 7, col = idx & 127;
            float r  = sigmoidf_(srz[row][col]);
            float zg = sigmoidf_(srz[row][HH + col]);
            float nv = tanhf(sinn[row][col] + r * (sgnn[row][col] + gbn[col]));
            float hn = nv + zg * (hreg[e] - nv);
            hreg[e] = hn;
            hf[row][col] = hn;
            hbf[row][col] = (__bf16)hn;
        }
        __syncthreads();
        // zx linear head: (16,128)x(128,8)
        if (tid < 128) {
            int bb = tid >> 3, o = tid & 7;
            float a = lb[o];
#pragma unroll 8
            for (int k = 0; k < HH; ++k) a += hf[bb][k] * lw[o * HH + k];
            zx_out[((size_t)(b0 + bb) * TT + t) * NZ + o] = a;
        }
        __syncthreads();
    }
}

// ---------------- kernel 2: zy GRU over L steps -> final hidden ----------------
__global__ __launch_bounds__(256) void zy_gru_kernel(
    const float* __restrict__ y_past, const float* __restrict__ zx,
    const float* __restrict__ tarr,
    const __bf16* __restrict__ wihT, const __bf16* __restrict__ whhT,
    const float* __restrict__ gb, const float* __restrict__ gbn,
    float* __restrict__ zy_cur)
{
    __shared__ __bf16 xs[16][32];
    __shared__ __bf16 hbf[16][HH];
    __shared__ float  hf[16][HH];
    __shared__ float  srz[16][2 * HH];
    __shared__ float  sinn[16][HH];
    __shared__ float  sgnn[16][HH];

    const int b0 = blockIdx.x * 16;
    const int tid = threadIdx.x;
    const int wv = tid >> 5, lane = tid & 31;
    const int nn = lane & 15, half = lane >> 4;

    v16bf bih[3], bhh[3][4];
    float bias[3];
#pragma unroll
    for (int j = 0; j < 3; ++j) {
        int n0 = wv * 16 + j * HH;
        bih[j] = load_b_frag(wihT, G3, lane, 0, n0);
#pragma unroll
        for (int ks = 0; ks < 4; ++ks) bhh[j][ks] = load_b_frag(whhT, G3, lane, ks * 32, n0);
        bias[j] = gb[n0 + nn];
    }

    for (int i = tid; i < 16 * HH; i += 256) {
        (&hf[0][0])[i] = 0.0f;
        (&hbf[0][0])[i] = (__bf16)0.0f;
    }
    float hreg[8];
#pragma unroll
    for (int e = 0; e < 8; ++e) hreg[e] = 0.0f;
    __syncthreads();

    for (int t = 0; t < LL; ++t) {
        // x tile: [y_past(8), zx(8), t, pad]
        for (int i = tid; i < 16 * 32; i += 256) {
            int r = i >> 5, c = i & 31;
            float v = 0.0f;
            if (c < 8)       v = y_past[((size_t)(b0 + r) * LL + t) * 8 + c];
            else if (c < 16) v = zx[((size_t)(b0 + r) * TT + t) * NZ + (c - 8)];
            else if (c == 16) v = tarr[t];
            (&xs[0][0])[i] = (__bf16)v;
        }
        __syncthreads();

        v16bf ax = load_a_frag(&xs[0][0], 32, lane, 0);
        v16bf ah[4];
#pragma unroll
        for (int ks = 0; ks < 4; ++ks) ah[ks] = load_a_frag(&hbf[0][0], HH, lane, ks * 32);

        v8f acc_i[3], acc_g[3];
#pragma unroll
        for (int j = 0; j < 3; ++j) {
            v8f zi = {};
            acc_i[j] = wmma_bf16(ax, bih[j], zi);
            v8f zg = {};
#pragma unroll
            for (int ks = 0; ks < 4; ++ks) zg = wmma_bf16(ah[ks], bhh[j][ks], zg);
            acc_g[j] = zg;
        }
#pragma unroll
        for (int v = 0; v < 8; ++v) {
            int row = v + (half << 3);
            int col = wv * 16 + nn;
            srz[row][col]      = acc_i[0][v] + acc_g[0][v] + bias[0];
            srz[row][HH + col] = acc_i[1][v] + acc_g[1][v] + bias[1];
            sinn[row][col]     = acc_i[2][v] + bias[2];
            sgnn[row][col]     = acc_g[2][v];
        }
        __syncthreads();
#pragma unroll
        for (int e = 0; e < 8; ++e) {
            int idx = tid + 256 * e;
            int row = idx >> 7, col = idx & 127;
            float r  = sigmoidf_(srz[row][col]);
            float zg = sigmoidf_(srz[row][HH + col]);
            float nv = tanhf(sinn[row][col] + r * (sgnn[row][col] + gbn[col]));
            float hn = nv + zg * (hreg[e] - nv);
            hreg[e] = hn;
            hf[row][col] = hn;
            hbf[row][col] = (__bf16)hn;
        }
        __syncthreads();
    }
    // write final hidden
    for (int i = tid; i < 16 * HH; i += 256) {
        int row = i >> 7, col = i & 127;
        zy_cur[(size_t)(b0 + row) * HH + col] = hf[row][col];
    }
}

// ---------------- kernel 3: RK4 CDE scan + readout ----------------
__global__ __launch_bounds__(256) void cde_rk4_kernel(
    const float* __restrict__ tarr,
    const __bf16* __restrict__ w1T, const __bf16* __restrict__ w2T,
    const __bf16* __restrict__ w3F,   // fragment-major packed f_w3
    const float* __restrict__ b1, const float* __restrict__ b2, const float* __restrict__ b3,
    const float* __restrict__ r_w, const float* __restrict__ r_b,
    const float* __restrict__ zx, const float* __restrict__ zy_cur,
    float* __restrict__ out)
{
    __shared__ __bf16 zin[16][HH];
    __shared__ __bf16 hb[16][HH];
    __shared__ __bf16 ob[16][HH * CF];   // 16 x 1152
    __shared__ float  zf[16][HH];
    __shared__ float  dxs[16][CF];

    const int b0 = blockIdx.x * 16;
    const int tid = threadIdx.x;
    const int wv = tid >> 5, lane = tid & 31;
    const int nn = lane & 15, half = lane >> 4;

    const v16bf* __restrict__ w3f = (const v16bf*)w3F;   // [(tile*4+ks)*32 + lane]

    // preload f_w1T / f_w2T fragments (wave wv owns N-tile wv of 128)
    v16bf fw1[4], fw2[4];
#pragma unroll
    for (int ks = 0; ks < 4; ++ks) {
        fw1[ks] = load_b_frag(w1T, HH, lane, ks * 32, wv * 16);
        fw2[ks] = load_b_frag(w2T, HH, lane, ks * 32, wv * 16);
    }
    const float bias1 = b1[wv * 16 + nn];
    const float bias2 = b2[wv * 16 + nn];
    float bias3[9];
#pragma unroll
    for (int j = 0; j < 9; ++j) bias3[j] = b3[(wv + 8 * j) * 16 + nn];

    // init z from zy_current
    float z[8], acc[8], kc[8];
#pragma unroll
    for (int e = 0; e < 8; ++e) {
        int idx = tid + 256 * e;
        int row = idx >> 7, col = idx & 127;
        z[e] = zy_cur[(size_t)(b0 + row) * HH + col];
        zf[row][col] = z[e];
        zin[row][col] = (__bf16)z[e];
    }
    __syncthreads();
    // y_hat[:, 0, :]
    if (tid < 128) {
        int bb = tid >> 3, o = tid & 7;
        float a = r_b[o];
#pragma unroll 8
        for (int k = 0; k < HH; ++k) a += zf[bb][k] * r_w[o * HH + k];
        out[((size_t)(b0 + bb) * FF + 0) * OC + o] = a;
    }
    __syncthreads();

    for (int s = 0; s < FF - 1; ++s) {
        // dX for this knot interval
        if (tid < 16 * CF) {
            int row = tid / CF, c = tid % CF;
            float d;
            if (c < NZ)
                d = zx[((size_t)(b0 + row) * TT + (LL + s + 1)) * NZ + c]
                  - zx[((size_t)(b0 + row) * TT + (LL + s)) * NZ + c];
            else
                d = tarr[LL + s + 1] - tarr[LL + s];
            dxs[row][c] = d;
        }
#pragma unroll
        for (int e = 0; e < 8; ++e) acc[e] = 0.0f;
        __syncthreads();

#pragma unroll 1
        for (int pass = 0; pass < 4; ++pass) {
            // --- GEMM1: h1 = softplus(zin @ w1T + b1) ---
            v16bf a0[4];
#pragma unroll
            for (int ks = 0; ks < 4; ++ks) a0[ks] = load_a_frag(&zin[0][0], HH, lane, ks * 32);
            v8f h8 = {};
#pragma unroll
            for (int ks = 0; ks < 4; ++ks) h8 = wmma_bf16(a0[ks], fw1[ks], h8);
#pragma unroll
            for (int v = 0; v < 8; ++v)
                hb[v + (half << 3)][wv * 16 + nn] = (__bf16)softplusf_(h8[v] + bias1);
            __syncthreads();

            // --- GEMM2: h2 = softplus(h1 @ w2T + b2) ---
            v16bf a1[4];
#pragma unroll
            for (int ks = 0; ks < 4; ++ks) a1[ks] = load_a_frag(&hb[0][0], HH, lane, ks * 32);
            v8f g8 = {};
#pragma unroll
            for (int ks = 0; ks < 4; ++ks) g8 = wmma_bf16(a1[ks], fw2[ks], g8);
            __syncthreads();
#pragma unroll
            for (int v = 0; v < 8; ++v)
                hb[v + (half << 3)][wv * 16 + nn] = (__bf16)softplusf_(g8[v] + bias2);
            __syncthreads();

            // --- GEMM3: o = tanh(h2 @ w3 + b3), N=1152 (9 tiles/wave) ---
            v16bf a2[4];
#pragma unroll
            for (int ks = 0; ks < 4; ++ks) a2[ks] = load_a_frag(&hb[0][0], HH, lane, ks * 32);
#pragma unroll 1
            for (int j = 0; j < 9; ++j) {
                int tIdx = wv + 8 * j;
                const v16bf* base = w3f + (size_t)tIdx * (4 * 32);
                if (j < 8)   // prefetch next tile's fragments for this lane
                    __builtin_prefetch((const void*)(w3f + (size_t)(tIdx + 8) * (4 * 32) + lane), 0, 1);
                v8f o8 = {};
#pragma unroll
                for (int ks = 0; ks < 4; ++ks) {
                    v16bf bw = base[ks * 32 + lane];
                    o8 = wmma_bf16(a2[ks], bw, o8);
                }
                int n0 = tIdx * 16;
#pragma unroll
                for (int v = 0; v < 8; ++v)
                    ob[v + (half << 3)][n0 + nn] = (__bf16)tanhf(o8[v] + bias3[j]);
            }
            __syncthreads();

            // --- contraction k[b,h] = sum_c o[b,h,c] * dx[b,c] ---
#pragma unroll
            for (int e = 0; e < 8; ++e) {
                int idx = tid + 256 * e;
                int row = idx >> 7, col = idx & 127;
                float sacc = 0.0f;
#pragma unroll
                for (int c = 0; c < CF; ++c)
                    sacc += (float)ob[row][col * CF + c] * dxs[row][c];
                kc[e] = sacc;
            }
            // --- RK4 bookkeeping + next probe state ---
            float wk = (pass == 0 || pass == 3) ? 1.0f : 2.0f;
            if (pass < 3) {
                float cz = (pass == 2) ? 1.0f : 0.5f;
#pragma unroll
                for (int e = 0; e < 8; ++e) {
                    acc[e] += wk * kc[e];
                    int idx = tid + 256 * e;
                    int row = idx >> 7, col = idx & 127;
                    zin[row][col] = (__bf16)(z[e] + cz * kc[e]);
                }
            } else {
#pragma unroll
                for (int e = 0; e < 8; ++e) {
                    acc[e] += kc[e];
                    z[e] += acc[e] * (1.0f / 6.0f);
                    int idx = tid + 256 * e;
                    int row = idx >> 7, col = idx & 127;
                    zin[row][col] = (__bf16)z[e];
                    zf[row][col] = z[e];
                }
            }
            __syncthreads();
        }

        // readout y_hat[:, s+1, :]
        if (tid < 128) {
            int bb = tid >> 3, o = tid & 7;
            float a = r_b[o];
#pragma unroll 8
            for (int k = 0; k < HH; ++k) a += zf[bb][k] * r_w[o * HH + k];
            out[((size_t)(b0 + bb) * FF + (s + 1)) * OC + o] = a;
        }
        __syncthreads();
    }
}

// ---------------- host launcher ----------------
extern "C" void kernel_launch(void* const* d_in, const int* in_sizes, int n_in,
                              void* d_out, int out_size, void* d_ws, size_t ws_size,
                              hipStream_t stream)
{
    (void)in_sizes; (void)n_in; (void)out_size; (void)ws_size;

    const float* y_past  = (const float*)d_in[0];
    const float* tarr    = (const float*)d_in[1];
    const float* coeffs  = (const float*)d_in[2];
    /* d_in[3] = input_length (int scalar) — constant 256 per reference */
    const float* zx_wih  = (const float*)d_in[4];
    const float* zx_whh  = (const float*)d_in[5];
    const float* zx_b    = (const float*)d_in[6];
    const float* zx_bn   = (const float*)d_in[7];
    const float* zxl_w   = (const float*)d_in[8];
    const float* zxl_b   = (const float*)d_in[9];
    const float* zy_wih  = (const float*)d_in[10];
    const float* zy_whh  = (const float*)d_in[11];
    const float* zy_b    = (const float*)d_in[12];
    const float* zy_bn   = (const float*)d_in[13];
    const float* f_w1    = (const float*)d_in[14];
    const float* f_b1    = (const float*)d_in[15];
    const float* f_w2    = (const float*)d_in[16];
    const float* f_b2    = (const float*)d_in[17];
    const float* f_w3    = (const float*)d_in[18];
    const float* f_b3    = (const float*)d_in[19];
    const float* r_w     = (const float*)d_in[20];
    const float* r_b     = (const float*)d_in[21];

    char* ws = (char*)d_ws;
    size_t off = 0;
    auto carve = [&](size_t bytes) { char* p = ws + off; off += (bytes + 255) & ~(size_t)255; return p; };

    __bf16* zx_wihT = (__bf16*)carve((size_t)32  * G3 * 2);
    __bf16* zx_whhT = (__bf16*)carve((size_t)HH  * G3 * 2);
    __bf16* zy_wihT = (__bf16*)carve((size_t)32  * G3 * 2);
    __bf16* zy_whhT = (__bf16*)carve((size_t)HH  * G3 * 2);
    __bf16* f_w1T   = (__bf16*)carve((size_t)HH  * WW * 2);
    __bf16* f_w2T   = (__bf16*)carve((size_t)WW  * WW * 2);
    __bf16* f_w3F   = (__bf16*)carve((size_t)(HH * CF) * WW * 2);   // fragment-major pack
    float*  zx_buf  = (float*) carve((size_t)BB * TT * NZ * 4);
    float*  zy_cur  = (float*) carve((size_t)BB * HH * 4);

    auto tp = [&](const float* src, __bf16* dst, int R, int K, int Kpad) {
        int total = Kpad * R;
        transpose_to_bf16<<<(total + 255) / 256, 256, 0, stream>>>(src, dst, R, K, Kpad);
    };
    tp(zx_wih, zx_wihT, G3, XC + 1, 32);
    tp(zx_whh, zx_whhT, G3, HH, HH);
    tp(zy_wih, zy_wihT, G3, 17, 32);
    tp(zy_whh, zy_whhT, G3, HH, HH);
    tp(f_w1,   f_w1T,   WW, HH, HH);
    tp(f_w2,   f_w2T,   WW, WW, WW);
    {
        int total = 72 * 4 * 32 * 16;
        pack_w3_frags<<<(total + 255) / 256, 256, 0, stream>>>(f_w3, f_w3F);
    }

    zx_gru_kernel<<<BB / 16, 256, 0, stream>>>(coeffs, tarr, zx_wihT, zx_whhT,
                                               zx_b, zx_bn, zxl_w, zxl_b, zx_buf);
    zy_gru_kernel<<<BB / 16, 256, 0, stream>>>(y_past, zx_buf, tarr, zy_wihT, zy_whhT,
                                               zy_b, zy_bn, zy_cur);
    cde_rk4_kernel<<<BB / 16, 256, 0, stream>>>(tarr, f_w1T, f_w2T, f_w3F,
                                                f_b1, f_b2, f_b3, r_w, r_b,
                                                zx_buf, zy_cur, (float*)d_out);
}
